// GraphAttentionLayer_43241730737064
// MI455X (gfx1250) — matrix operand order repaired and checked by reading, not compiled, not run on hardware
//
#include <hip/hip_runtime.h>
#include <hip/hip_bf16.h>

// GAT layer, MI455X (gfx1250, wave32).
// softmax over j of (s1[i]+s2[j]+b) == softmax(s2)  =>  output = adj @ (p ⊙ h).
// Dominant cost: adj stream (256 MB @ 23.3 TB/s ~= 11 us floor). adj tiles are
// DMA'd into LDS by the Tensor Data Mover (double-buffered, TENSORcnt-gated),
// converted to bf16 on read, and fed to V_WMMA_F32_16X16X32_BF16 (f32 accum).
// B fragments (hT in L2) are software-pipelined one K-step ahead in registers
// so WMMAs never wait on loads issued in the same step.

typedef __bf16 bf16_t;
typedef __attribute__((ext_vector_type(16))) __bf16 bf16x16;
typedef __attribute__((ext_vector_type(8)))  float  f32x8;
typedef __attribute__((ext_vector_type(4)))  unsigned int u32x4;
typedef __attribute__((ext_vector_type(8)))  int  i32x8;
typedef __attribute__((ext_vector_type(4)))  int  i32x4;

#define NN   8192   // nodes
#define FIN  512
#define FOUT 256
#define KC   128            // K-chunk per TDM tile (f32 dwords per row)
#define LPITCH (KC + 4)     // LDS row pitch in dwords (TDM pad: 4 dwords / 128)

// ---------------- TDM: DMA a 32 x KC f32 tile (row stride K) into LDS -------
// D# per CDNA5 ISA ch.8. pad_interval=128 dwords, pad_amount=4 dwords ->
// LDS pitch 132 dwords (bank-conflict mitigation for the ds_load_b128 reads).
__device__ inline void tdm_load_tile_f32(unsigned lds_off, const float* gsrc,
                                         unsigned K) {
  unsigned long long ga = (unsigned long long)(uintptr_t)gsrc;
  u32x4 g0;
  g0[0] = 1u;                                   // count=1, user descriptor
  g0[1] = lds_off;                              // lds_addr (bytes)
  g0[2] = (unsigned)(ga & 0xffffffffu);         // global_addr[31:0]
  g0[3] = (unsigned)(ga >> 32) | (2u << 30);    // global_addr[56:32] | type=2
  i32x8 g1;
  g1[0] = (int)((2u << 16)                      // data_size = 4 bytes
              | (1u << 20)                      // pad_enable
              | (6u << 22)                      // pad_interval: every 128 dwords
              | (3u << 25));                    // pad_amount: 4 dwords
  g1[1] = (int)(K << 16);                       // tensor_dim0[15:0] (row length)
  g1[2] = (int)((K >> 16) | (32u << 16));       // tensor_dim0 hi | tensor_dim1=32
  g1[3] = (int)(KC << 16);                      // tile_dim0 = KC
  g1[4] = (int)32;                              // tile_dim1 = 32 rows, tile_dim2=0
  g1[5] = (int)K;                               // tensor_dim0_stride lo
  g1[6] = 0;                                    // stride hi | dim1_stride lo
  g1[7] = 0;
  i32x4 z4 = {0, 0, 0, 0};
#if __has_include(<hip/amd_detail/amd_gfx1250_TDM.h>)
  i32x8 z8 = {0, 0, 0, 0, 0, 0, 0, 0};
  __builtin_amdgcn_tensor_load_to_lds(g0, g1, z4, z4, z8, 0);
#else
  __builtin_amdgcn_tensor_load_to_lds(g0, g1, z4, z4, 0);
#endif
}

// ---------------- WMMA fragment loaders (V_WMMA_F32_16X16X32_BF16) ----------

// A 16x32 bf16 fragment from the LDS f32 tile (cvt on read).
// lane l: m=l&15, half=l>>4; elems 0..15 = A[m][ks+16h .. +15].
__device__ inline bf16x16 lds_a_frag(const float* __restrict__ arow, int ks, int lane) {
  const float* p = arow + ks + 16 * (lane >> 4);
  float4 q0 = *(const float4*)(p);
  float4 q1 = *(const float4*)(p + 4);
  float4 q2 = *(const float4*)(p + 8);
  float4 q3 = *(const float4*)(p + 12);
  bf16x16 a;
  a[0]  = (bf16_t)q0.x; a[1]  = (bf16_t)q0.y; a[2]  = (bf16_t)q0.z; a[3]  = (bf16_t)q0.w;
  a[4]  = (bf16_t)q1.x; a[5]  = (bf16_t)q1.y; a[6]  = (bf16_t)q1.z; a[7]  = (bf16_t)q1.w;
  a[8]  = (bf16_t)q2.x; a[9]  = (bf16_t)q2.y; a[10] = (bf16_t)q2.z; a[11] = (bf16_t)q2.w;
  a[12] = (bf16_t)q3.x; a[13] = (bf16_t)q3.y; a[14] = (bf16_t)q3.z; a[15] = (bf16_t)q3.w;
  return a;
}

// B 32x16 fragment from Bt = B^T, row-major [N][K] bf16 (contiguous 32B/lane).
__device__ inline bf16x16 load_b_frag(const bf16_t* __restrict__ Bt, int K,
                                      int n0, int k0, int lane) {
  const bf16_t* p = Bt + (size_t)(n0 + (lane & 15)) * K + k0 + 16 * (lane >> 4);
  return *(const bf16x16*)p;
}

// ---------------- GEMM: C[M][N] = A(f32) x Bt^T (+bias), TDM-pipelined ------
// 256 threads = 8 waves: 2(M) x 4(N); wave tile 16x64 (4 accumulators).
// adj/x tile (32 x KC f32) double-buffered in LDS via tensor_load_to_lds;
// B fragments double-buffered in registers one K-step ahead.
__global__ __launch_bounds__(256) void gat_gemm_bf16_tdm(
    const float* __restrict__ A, const bf16_t* __restrict__ Bt,
    const float* __restrict__ bias, float* __restrict__ C,
    int M, int K, int N) {
  __shared__ float Abuf[2][32][LPITCH];

  const int lane = threadIdx.x & 31;
  const int wave = threadIdx.x >> 5;
  const int mi = wave >> 2;            // 0..1
  const int ni = wave & 3;             // 0..3
  const int m0 = blockIdx.x * 32 + mi * 16;
  const int n0 = ni * 64;

  const float* gA = A + (size_t)(blockIdx.x * 32) * K;  // 32-row slab of A
  const float* arow = &Abuf[0][mi * 16 + (lane & 15)][0];

  f32x8 acc0 = {}, acc1 = {}, acc2 = {}, acc3 = {};

  if (wave == 0)   // prime the DMA pipeline: tile for k0 = 0 into buffer 0
    tdm_load_tile_f32((unsigned)(uintptr_t)&Abuf[0][0][0], gA, (unsigned)K);

  // prime the B register pipeline (fragments for kk = 0)
  bf16x16 c0 = load_b_frag(Bt, K, n0 +  0, 0, lane);
  bf16x16 c1 = load_b_frag(Bt, K, n0 + 16, 0, lane);
  bf16x16 c2 = load_b_frag(Bt, K, n0 + 32, 0, lane);
  bf16x16 c3 = load_b_frag(Bt, K, n0 + 48, 0, lane);

  int pb = 0;
  for (int k0 = 0; k0 < K; k0 += KC) {
    if (wave == 0) {
      if (k0 + KC < K) {               // issue next tile into the other buffer
        tdm_load_tile_f32((unsigned)(uintptr_t)&Abuf[pb ^ 1][0][0],
                          gA + k0 + KC, (unsigned)K);
        __builtin_amdgcn_s_wait_tensorcnt(1);   // current tile landed
      } else {
        __builtin_amdgcn_s_wait_tensorcnt(0);
      }
    }
    __syncthreads();                   // tile pb visible to all waves

    const float* ab = arow + (size_t)pb * 32 * LPITCH;
#pragma unroll
    for (int ks = 0; ks < KC; ks += 32) {
      // issue NEXT step's B loads first (wrap to 0 on the final step: branchless
      // tail, 4 redundant loads per block, keeps the pipeline schedulable)
      const int kraw = k0 + ks + 32;
      const int kn = (kraw < K) ? kraw : 0;
      bf16x16 d0 = load_b_frag(Bt, K, n0 +  0, kn, lane);
      bf16x16 d1 = load_b_frag(Bt, K, n0 + 16, kn, lane);
      bf16x16 d2 = load_b_frag(Bt, K, n0 + 32, kn, lane);
      bf16x16 d3 = load_b_frag(Bt, K, n0 + 48, kn, lane);
      __builtin_prefetch(Bt + (size_t)(n0 + (lane & 15)) * K + kn + 512, 0, 1);

      // A fragment from LDS (short ds latency), then WMMAs on CURRENT B regs
      bf16x16 a = lds_a_frag(ab, ks, lane);
      acc0 = __builtin_amdgcn_wmma_f32_16x16x32_bf16(false, a, false, c0, (short)0, acc0, false, false);
      acc1 = __builtin_amdgcn_wmma_f32_16x16x32_bf16(false, a, false, c1, (short)0, acc1, false, false);
      acc2 = __builtin_amdgcn_wmma_f32_16x16x32_bf16(false, a, false, c2, (short)0, acc2, false, false);
      acc3 = __builtin_amdgcn_wmma_f32_16x16x32_bf16(false, a, false, c3, (short)0, acc3, false, false);
      c0 = d0; c1 = d1; c2 = d2; c3 = d3;
    }

    __syncthreads();                   // everyone done with tile pb before reuse
    pb ^= 1;
  }

  // C/D layout: elem r of lane l -> row m0 + 8*(l>>4) + r, col n0 + 16*t + (l&15)
  const int half = lane >> 4, nl = lane & 15;
  f32x8 accs[4] = {acc0, acc1, acc2, acc3};
#pragma unroll
  for (int t = 0; t < 4; ++t) {
    const int col = n0 + t * 16 + nl;
    const float bb = bias ? bias[col] : 0.0f;
#pragma unroll
    for (int r = 0; r < 8; ++r)
      C[(size_t)(m0 + 8 * half + r) * N + col] = accs[t][r] + bb;
  }
}

// ---------------- small prep / reduction kernels ----------------------------

__global__ __launch_bounds__(256) void gat_convert_w(const float* __restrict__ W,
                                                     bf16_t* __restrict__ Wt) {
  int idx = blockIdx.x * 256 + threadIdx.x;     // < FIN*FOUT
  int k = idx >> 8, n = idx & (FOUT - 1);
  Wt[(size_t)n * FIN + k] = (bf16_t)W[idx];     // Wt[n][k] = W[k][n]
}

__global__ __launch_bounds__(256) void gat_rowdot(const float* __restrict__ h,
                                                  const float* __restrict__ a_w,
                                                  float* __restrict__ s2) {
  __shared__ float red[256];
  const int i = blockIdx.x, t = threadIdx.x;
  red[t] = h[(size_t)i * FOUT + t] * a_w[FOUT + t];
  __syncthreads();
#pragma unroll
  for (int s = 128; s > 0; s >>= 1) {
    if (t < s) red[t] += red[t + s];
    __syncthreads();
  }
  if (t == 0) s2[i] = red[0];
}

__global__ __launch_bounds__(1024) void gat_softmax_stats(const float* __restrict__ s2,
                                                          float* __restrict__ stats) {
  __shared__ float red[1024];
  const int t = threadIdx.x;
  float m = -1e30f;
  for (int j = t; j < NN; j += 1024) m = fmaxf(m, s2[j]);
  red[t] = m; __syncthreads();
#pragma unroll
  for (int s = 512; s > 0; s >>= 1) {
    if (t < s) red[t] = fmaxf(red[t], red[t + s]);
    __syncthreads();
  }
  const float mm = red[0];
  __syncthreads();
  float z = 0.0f;
  for (int j = t; j < NN; j += 1024) z += __expf(s2[j] - mm);
  red[t] = z; __syncthreads();
#pragma unroll
  for (int s = 512; s > 0; s >>= 1) {
    if (t < s) red[t] += red[t + s];
    __syncthreads();
  }
  if (t == 0) { stats[0] = mm; stats[1] = 1.0f / red[0]; }
}

__global__ __launch_bounds__(256) void gat_scale_transpose(
    const float* __restrict__ h, const float* __restrict__ s2,
    const float* __restrict__ stats, bf16_t* __restrict__ hT) {
  const int j = blockIdx.x * 256 + threadIdx.x;  // node
  const int n = blockIdx.y;                       // feature
  const float p = __expf(s2[j] - stats[0]) * stats[1];
  hT[(size_t)n * NN + j] = (bf16_t)(p * h[(size_t)j * FOUT + n]);
}

// ---------------- launcher --------------------------------------------------
extern "C" void kernel_launch(void* const* d_in, const int* in_sizes, int n_in,
                              void* d_out, int out_size, void* d_ws, size_t ws_size,
                              hipStream_t stream) {
  const float* x   = (const float*)d_in[0];   // (8192, 512)
  const float* adj = (const float*)d_in[1];   // (8192, 8192)
  const float* W_w = (const float*)d_in[2];   // (512, 256)
  const float* W_b = (const float*)d_in[3];   // (256,)
  const float* a_w = (const float*)d_in[4];   // (512, 1); a_b cancels in softmax
  float* out = (float*)d_out;                 // (8192, 256)

  // workspace layout (bytes), all 32B-aligned
  char* ws = (char*)d_ws;
  float*  h     = (float*) (ws);                             //  8 MB  f32 h
  bf16_t* Wt    = (bf16_t*)(ws + 8388608);                   // 256 KB W^T bf16
  bf16_t* hT    = (bf16_t*)(ws + 8650752);                   //  4 MB  (p⊙h)^T bf16
  float*  s2    = (float*) (ws + 12845056);                  // 32 KB
  float*  stats = (float*) (ws + 12877824);                  // 8 B

  gat_convert_w<<<(FIN * FOUT) / 256, 256, 0, stream>>>(W_w, Wt);
  // h = x @ W + b   (M=8192, K=512, N=256)
  gat_gemm_bf16_tdm<<<NN / 32, 256, 0, stream>>>(x, Wt, W_b, h, NN, FIN, FOUT);
  gat_rowdot<<<NN, 256, 0, stream>>>(h, a_w, s2);
  gat_softmax_stats<<<1, 1024, 0, stream>>>(s2, stats);
  gat_scale_transpose<<<dim3(NN / 256, FOUT), 256, 0, stream>>>(h, s2, stats, hT);
  // out = adj @ (p⊙h)   (M=8192, K=8192, N=256) — TDM streams adj, HBM-bound
  gat_gemm_bf16_tdm<<<NN / 32, 256, 0, stream>>>(adj, hT, nullptr, out, NN, NN, FOUT);
}
